// DGCRNNCell_21792664060192
// MI455X (gfx1250) — matrix operand order repaired and checked by reading, not compiled, not run on hardware
//
#include <hip/hip_runtime.h>

// ---------------- problem constants ----------------
#define BB   512      // batch
#define NN   199      // nodes
#define FF   64       // features (=units U)
#define HH   4        // heads
#define NP   224      // N padded to multiple of 32 (WMMA K chunk)
#define NK   (NP/32)  // 7 K-chunks over the node dimension
#define NT   (NP/16)  // 14 16-wide tiles over the node dimension
#define OD   455      // output last dim = H*U + N = 256 + 199
#define NEGC 1e16f

typedef __attribute__((ext_vector_type(16))) __bf16 v16bf;
typedef __attribute__((ext_vector_type(8)))  __bf16 v8bf;
typedef __attribute__((ext_vector_type(8)))  float  v8f;

// ---------------- workspace layout (bytes) ----------------
// AWt : bf16 [H][NP][NP]   (A-matrix form of adjacency*W, transposed)
// ADD : f32  [H][NP][NP]   (bias1[j] - 1e16*(1-A[i][j]), padded -> -1e16)
// XT  : bf16 [B][F][NP]    (X transposed per batch, zero padded)
// kT  : bf16 [H][NP][F]    (kernel transposed)
// TT  : bf16 [H][F][F]     (T transposed)
#define OFF_AWT 0
#define SZ_AWT  ((size_t)HH*NP*NP*2)
#define OFF_ADD (OFF_AWT + SZ_AWT)
#define SZ_ADD  ((size_t)HH*NP*NP*4)
#define OFF_XT  (OFF_ADD + SZ_ADD)
#define SZ_XT   ((size_t)BB*FF*NP*2)
#define OFF_KT  (OFF_XT + SZ_XT)
#define SZ_KT   ((size_t)HH*NP*FF*2)
#define OFF_TT  (OFF_KT + SZ_KT)
#define SZ_TT   ((size_t)HH*FF*FF*2)

// ---------------- WMMA helpers ----------------
__device__ inline v8f wmma_bf16(v16bf a, v16bf b, v8f c) {
  // v_wmma_f32_16x16x32_bf16, D = A x B + C, f32 accumulate
  return __builtin_amdgcn_wmma_f32_16x16x32_bf16(
      /*neg_a=*/false, a, /*neg_b=*/false, b,
      /*c_mod=*/(short)0, c, /*reuse_a=*/false, /*reuse_b=*/false);
}

// A-matrix tile loader (16x32 bf16). Per ISA layout:
// lanes 0-15 : row = rowBase+lane,  regs 0-3 hold K 0..7,  regs 4-7 hold K 16..23
// lanes 16-31: row = rowBase+lane-16, regs 0-3 hold K 8..15, regs 4-7 hold K 24..31
__device__ inline v16bf load_a(const __bf16* p, int stride, int rowBase, int kb, int lane) {
  int r   = rowBase + (lane & 15);
  int sel = (lane >> 4) & 1;
  const __bf16* q = p + (size_t)r * stride + kb + sel * 8;
  v8bf lo = *(const v8bf*)q;          // K kb+sel*8    .. +7
  v8bf hi = *(const v8bf*)(q + 16);   // K kb+16+sel*8 .. +7
  return __builtin_shufflevector(lo, hi, 0,1,2,3,4,5,6,7,8,9,10,11,12,13,14,15);
}

// B-matrix tile loader (32x16 bf16), reading a PRE-TRANSPOSED operand Bt[col][k]
// (K contiguous). Same K order as load_a: lane holds column n = lane&15 with
// K 0..7,16..23 (lanes 0-15) or K 8..15,24..31 (lanes 16-31).
__device__ inline v16bf load_b(const __bf16* p, int stride, int colBase, int kb, int lane) {
  int c   = colBase + (lane & 15);
  int sel = (lane >> 4) & 1;
  const __bf16* q = p + (size_t)c * stride + kb + sel * 8;
  v8bf lo = *(const v8bf*)q;
  v8bf hi = *(const v8bf*)(q + 16);
  return __builtin_shufflevector(lo, hi, 0,1,2,3,4,5,6,7,8,9,10,11,12,13,14,15);
}

// ---------------- prep kernels (operand conversion / transpose / pad) ----------------
__global__ void prep_awt(const float* __restrict__ A, const float* __restrict__ W,
                         const float* __restrict__ bias1,
                         __bf16* __restrict__ AWt, float* __restrict__ ADD) {
  int idx = blockIdx.x * blockDim.x + threadIdx.x;
  if (idx >= HH * NP * NP) return;
  int n = idx % NP; int t = idx / NP; int m = t % NP; int h = t / NP;
  // AWt[h][m][n] = A[n][m] * W[h][n][m]   (transposed so feat = AWt rows x X)
  float aw = 0.f;
  if (m < NN && n < NN) aw = A[n * NN + m] * W[((size_t)h * NN + n) * NN + m];
  AWt[idx] = (__bf16)aw;
  // ADD[h][i][j] = bias1[h][j] - NEG*(1-A[i][j]); padding fully masked
  float ad = -NEGC;
  if (m < NN && n < NN) ad = bias1[h * NN + n] - NEGC * (1.0f - A[m * NN + n]);
  ADD[idx] = ad;
}

__global__ void prep_xt(const float* __restrict__ X, __bf16* __restrict__ XT) {
  int idx = blockIdx.x * blockDim.x + threadIdx.x;
  if (idx >= BB * FF * NP) return;
  int n = idx % NP; int t = idx / NP; int f = t % FF; int b = t / FF;
  float v = (n < NN) ? X[((size_t)b * NN + n) * FF + f] : 0.f;
  XT[idx] = (__bf16)v;
}

__global__ void prep_kt(const float* __restrict__ K, __bf16* __restrict__ kT) {
  int idx = blockIdx.x * blockDim.x + threadIdx.x;
  if (idx >= HH * NP * FF) return;
  int f = idx % FF; int t = idx / FF; int j = t % NP; int h = t / NP;
  float v = (j < NN) ? K[((size_t)h * FF + f) * NN + j] : 0.f;
  kT[idx] = (__bf16)v;
}

__global__ void prep_tt(const float* __restrict__ T, __bf16* __restrict__ TT) {
  int idx = blockIdx.x * blockDim.x + threadIdx.x;
  if (idx >= HH * FF * FF) return;
  int f = idx % FF; int t = idx / FF; int u = t % FF; int h = t / FF;
  TT[idx] = (__bf16)T[((size_t)h * FF + f) * FF + u];
}

// ---------------- fused DynamicGC kernel ----------------
// One block per (batch, head); 4 waves; each wave owns 16-row tiles end-to-end:
// GEMM1 -> feat(LDS) -> GEMM2 (dense in regs) -> masked softmax (in regs,
// shfl row-reduce) -> mask(LDS, bf16 A-layout) -> GEMM3 -> node(LDS) -> GEMM4.
// A-tiles for each GEMM are loaded as one burst ahead of the WMMA burst so a
// single (partial) loadcnt/dscnt wait overlaps load latency with WMMA issue.
__global__ __launch_bounds__(128) void dgc_fused(
    const __bf16* __restrict__ AWt, const float* __restrict__ ADD,
    const __bf16* __restrict__ XT,  const __bf16* __restrict__ kT,
    const __bf16* __restrict__ TT,  const float* __restrict__ bias2,
    float* __restrict__ out) {
  __shared__ __align__(16) __bf16 lds_mask[4][16 * NP];  // per-wave mask tile
  __shared__ __align__(16) __bf16 lds_fn[4][16 * FF];    // per-wave feat/node tile

  const int blk  = blockIdx.x;
  const int b    = blk >> 2;          // / HH
  const int h    = blk & 3;           // % HH
  const int tid  = threadIdx.x;
  const int wave = tid >> 5;
  const int lane = tid & 31;
  const int l15  = lane & 15;
  const int hi8  = ((lane >> 4) & 1) * 8;

  const __bf16* AWth = AWt + (size_t)h * NP * NP;
  const float*  ADDh = ADD + (size_t)h * NP * NP;
  const __bf16* XTb  = XT  + (size_t)b * FF * NP;
  const __bf16* kTh  = kT  + (size_t)h * NP * FF;
  const __bf16* TTh  = TT  + (size_t)h * FF * FF;
  float* outb = out + (size_t)b * NN * OD;

  __bf16* mymask = lds_mask[wave];
  __bf16* myfn   = lds_fn[wave];

  for (int rt = wave; rt < NT; rt += 4) {
    const int m0 = rt * 16;

    // prefetch next iteration's AW rows into near cache (global_prefetch_b8)
    if (rt + 4 < NT) {
      const __bf16* nxt = AWth + (size_t)(m0 + 64 + l15) * NP + hi8 * 16;
      __builtin_prefetch(nxt, 0, 1);
    }

    // ---- GEMM1: feat(16xF) = AWt[m0..m0+15][:] x X ----
    // burst-load all 7 A-tiles, then run the 28-WMMA burst
    v16bf a1[NK];
#pragma unroll
    for (int k = 0; k < NK; ++k) a1[k] = load_a(AWth, NP, m0, k * 32, lane);
    v8f fc[4] = {v8f{}, v8f{}, v8f{}, v8f{}};
#pragma unroll
    for (int k = 0; k < NK; ++k)
#pragma unroll
      for (int ct = 0; ct < 4; ++ct)
        fc[ct] = wmma_bf16(a1[k], load_b(XTb, NP, ct * 16, k * 32, lane), fc[ct]);

    // stage feat to LDS (C-layout -> row-major bf16)
#pragma unroll
    for (int ct = 0; ct < 4; ++ct)
#pragma unroll
      for (int r = 0; r < 8; ++r)
        myfn[(r + hi8) * FF + ct * 16 + l15] = (__bf16)fc[ct][r];

    // ---- GEMM2: dense(16xNP) = feat x kernel, all 14 col tiles in regs ----
    v16bf fa0 = load_a(myfn, FF, 0, 0, lane);
    v16bf fa1 = load_a(myfn, FF, 0, 32, lane);
    v8f dc[NT];
#pragma unroll
    for (int ct = 0; ct < NT; ++ct) {
      v8f c = {};
      c = wmma_bf16(fa0, load_b(kTh, FF, ct * 16, 0, lane), c);
      c = wmma_bf16(fa1, load_b(kTh, FF, ct * 16, 32, lane), c);
      dc[ct] = c;
    }

    // ---- masked softmax over each row (row = lanes 0-15 or 16-31 of a reg) ----
#pragma unroll
    for (int r = 0; r < 8; ++r) {
      const int row = m0 + r + hi8;
      float x[NT];
      float mx = -3.4e38f;
#pragma unroll
      for (int ct = 0; ct < NT; ++ct) {
        float v = dc[ct][r] + ADDh[(size_t)row * NP + ct * 16 + l15];
        x[ct] = v;
        mx = fmaxf(mx, v);
      }
      for (int s = 1; s < 16; s <<= 1) mx = fmaxf(mx, __shfl_xor(mx, s, 32));
      float ssum = 0.f;
#pragma unroll
      for (int ct = 0; ct < NT; ++ct) {
        float e = __expf(x[ct] - mx);
        x[ct] = e;
        ssum += e;
      }
      for (int s = 1; s < 16; s <<= 1) ssum += __shfl_xor(ssum, s, 32);
      const float inv = 1.f / ssum;
#pragma unroll
      for (int ct = 0; ct < NT; ++ct) {
        float p = x[ct] * inv;
        mymask[(r + hi8) * NP + ct * 16 + l15] = (__bf16)p;
        if (h == HH - 1) {  // reference concatenates the LAST head's mask
          int col = ct * 16 + l15;
          if (row < NN && col < NN)
            outb[(size_t)row * OD + HH * FF + col] = p;
        }
      }
    }

    // ---- GEMM3: node(16xF) = mask x X ----
    // burst-load the 7 mask A-tiles from LDS, then the 28-WMMA burst
    v16bf a3[NK];
#pragma unroll
    for (int k = 0; k < NK; ++k) a3[k] = load_a(mymask, NP, 0, k * 32, lane);
    v8f nc[4] = {v8f{}, v8f{}, v8f{}, v8f{}};
#pragma unroll
    for (int k = 0; k < NK; ++k)
#pragma unroll
      for (int ct = 0; ct < 4; ++ct)
        nc[ct] = wmma_bf16(a3[k], load_b(XTb, NP, ct * 16, k * 32, lane), nc[ct]);

#pragma unroll
    for (int ct = 0; ct < 4; ++ct)
#pragma unroll
      for (int r = 0; r < 8; ++r)
        myfn[(r + hi8) * FF + ct * 16 + l15] = (__bf16)nc[ct][r];

    // ---- GEMM4: out(16xU) = node x T, + bias2, store ----
    v16bf na0 = load_a(myfn, FF, 0, 0, lane);
    v16bf na1 = load_a(myfn, FF, 0, 32, lane);
#pragma unroll
    for (int ct = 0; ct < 4; ++ct) {
      v8f c = {};
      c = wmma_bf16(na0, load_b(TTh, FF, ct * 16, 0, lane), c);
      c = wmma_bf16(na1, load_b(TTh, FF, ct * 16, 32, lane), c);
      const float bb = bias2[h * FF + ct * 16 + l15];
#pragma unroll
      for (int r = 0; r < 8; ++r) {
        int row = m0 + r + hi8;
        if (row < NN)
          outb[(size_t)row * OD + h * FF + ct * 16 + l15] = c[r] + bb;
      }
    }
  }
}

// ---------------- launch ----------------
extern "C" void kernel_launch(void* const* d_in, const int* in_sizes, int n_in,
                              void* d_out, int out_size, void* d_ws, size_t ws_size,
                              hipStream_t stream) {
  (void)in_sizes; (void)n_in; (void)out_size; (void)ws_size;
  const float* X     = (const float*)d_in[0];
  const float* A     = (const float*)d_in[1];
  const float* W     = (const float*)d_in[2];
  const float* Kern  = (const float*)d_in[3];
  const float* T     = (const float*)d_in[4];
  const float* bias1 = (const float*)d_in[5];
  const float* bias2 = (const float*)d_in[6];
  float* out = (float*)d_out;

  char* ws = (char*)d_ws;
  __bf16* AWt = (__bf16*)(ws + OFF_AWT);
  float*  ADD = (float*)(ws + OFF_ADD);
  __bf16* XT  = (__bf16*)(ws + OFF_XT);
  __bf16* kT  = (__bf16*)(ws + OFF_KT);
  __bf16* TT  = (__bf16*)(ws + OFF_TT);

  const int tpb = 256;
  prep_awt<<<(HH * NP * NP + tpb - 1) / tpb, tpb, 0, stream>>>(A, W, bias1, AWt, ADD);
  prep_xt <<<(BB * FF * NP + tpb - 1) / tpb, tpb, 0, stream>>>(X, XT);
  prep_kt <<<(HH * NP * FF + tpb - 1) / tpb, tpb, 0, stream>>>(Kern, kT);
  prep_tt <<<(HH * FF * FF + tpb - 1) / tpb, tpb, 0, stream>>>(T, TT);

  dgc_fused<<<BB * HH, 128, 0, stream>>>(AWt, ADD, XT, kT, TT, bias2, out);
}
